// MHA_47425028883031
// MI455X (gfx1250) — matrix operand order repaired and checked by reading, not compiled, not run on hardware
//
// MHA forward for MI455X (gfx1250), wave32 + WMMA f16 (hi/lo split for ~fp32 accuracy),
// with gfx1250 async Global->LDS staging (ASYNCcnt) + double buffering in the GEMMs.
//
// Pipeline:
//   1) split x, wq, wk, wv, wo into (hi,lo) f16 pairs          (elementwise)
//   2) Q = x@wq^T, K = x@wk^T  -> split f16, layout [b,h,s,128]  (WMMA GEMM, async-LDS)
//      Vt = (x@wv^T)^T          -> split f16, layout [b,h,128,s]
//   3) fused attention per (b,h,16-row block):
//        S = Q K^T / sqrt(128) + bias + mask   (v_wmma_f32_16x16x32_f16 x3 per K-chunk)
//        softmax rows in LDS (hi/lo f16 storage), nontemporal-store score to d_out
//        ctx = P V  (split WMMA, V pre-transposed), store ctx pre-split [b,s,h*128]
//   4) out = ctx @ wo^T  (WMMA GEMM, async-LDS, f32 store)
//
// Roofline: ~275 GFLOP vs ~2.2 GB mandatory HBM traffic (score write + bias read)
// => memory-bound (~90us floor at 23.3 TB/s) once matmuls run on the f16 WMMA pipe;
// async double-buffered staging keeps the WMMA pipe fed in the GEMM phases.

#include <hip/hip_runtime.h>
#include <hip/hip_bf16.h>

typedef _Float16 f16;
typedef __attribute__((ext_vector_type(16))) _Float16 v16h;
typedef __attribute__((ext_vector_type(8)))  float    v8f;

#define S_LEN 2048
#define HEADS 16
#define DH    128
#define BATCH 4

union Frag16 { v16h v; float4 f4[2]; };

__device__ __forceinline__ v16h load_frag_2x8(const f16* p0, const f16* p1) {
  Frag16 u;
  u.f4[0] = *(const float4*)p0;   // halves [0..7]
  u.f4[1] = *(const float4*)p1;   // halves [8..15]
  return u.v;
}
__device__ __forceinline__ v16h load_frag_16(const f16* p) {
  Frag16 u;
  u.f4[0] = *(const float4*)p;
  u.f4[1] = *(const float4*)(p + 8);
  return u.v;
}

__device__ __forceinline__ v8f wmma16(v16h a, v16h b, v8f c) {
  // D = A(16x32 f16) * B(32x16 f16) + C(16x16 f32)
  return __builtin_amdgcn_wmma_f32_16x16x32_f16(false, a, false, b, (short)0, c, false, false);
}

// ---- gfx1250 async Global->LDS (ASYNCcnt) -----------------------------------
// Flat-address rule (ISA 10.2): LDS uses only the low 32 address bits, so the low 32 bits of a
// generic __shared__ pointer are the LDS byte offset the instruction wants.
__device__ __forceinline__ void async_load_b128(const void* lds_dst, const void* gaddr) {
  const uint32_t lds_off = (uint32_t)(uintptr_t)lds_dst;
  asm volatile("global_load_async_to_lds_b128 %0, %1, off"
               :: "v"(lds_off), "v"(gaddr)
               : "memory");
}
__device__ __forceinline__ void wait_async0() {
  asm volatile("s_wait_asynccnt 0" ::: "memory");
}

// ---------------------------------------------------------------------------
// Elementwise fp32 -> (hi,lo) f16 split:  v ~= hi + lo, products exact in f32.
// ---------------------------------------------------------------------------
__global__ __launch_bounds__(256) void split_f32_f16(
    const float* __restrict__ in, f16* __restrict__ oh, f16* __restrict__ ol, int n) {
  int i = blockIdx.x * 256 + threadIdx.x;
  if (i < n) {
    float v = in[i];
    f16 h = (f16)v;
    oh[i] = h;
    ol[i] = (f16)(v - (float)h);
  }
}

// ---------------------------------------------------------------------------
// Split-f16 WMMA GEMM:  C[M,N] = A[M,K] @ B[N,K]^T  (A,B as hi/lo f16 pairs)
// Block tile 64x64, 256 threads (8 waves), waves in 2x4 grid, 2 row-tiles each.
// Async double-buffered LDS staging: chunk k+1 streams in while chunk k computes.
// MODE 0: C fp32 row-major [M,N]
// MODE 1: split-f16 store, QKV layout  [b,h,s,128]   (row=b*S+s, col=h*128+c)
// MODE 2: split-f16 store, transposed  [b,h,128,s]
// ---------------------------------------------------------------------------
template <int MODE>
__global__ __launch_bounds__(256) void gemm_split(
    const f16* __restrict__ Ah, const f16* __restrict__ Al,
    const f16* __restrict__ Bh, const f16* __restrict__ Bl,
    int K, int N,
    float* __restrict__ Cf, f16* __restrict__ Ch, f16* __restrict__ Cl) {
  constexpr int LS = 40;        // padded LDS row stride (halves) for a 32-wide K chunk
  constexpr int HB = 64 * LS;   // halves per array per buffer
  __shared__ f16 sAh[2][HB], sAl[2][HB], sBh[2][HB], sBl[2][HB];  // 40 KB total

  const int tid  = threadIdx.x;
  const int wave = tid >> 5, lane = tid & 31;
  const int hi = lane >> 4, lm = lane & 15;
  const int wm = wave >> 2, wn = wave & 3;
  const int bm = blockIdx.y * 64, bn = blockIdx.x * 64;

  // staging geometry: one b128 (8 halves) per thread per array per chunk
  const int sr = tid >> 2, sc8 = (tid & 3) * 8;
  const size_t gA = (size_t)(bm + sr) * K + sc8;
  const size_t gB = (size_t)(bn + sr) * K + sc8;
  const int so = sr * LS + sc8;

  v8f acc[2] = {};

  // prologue: stage chunk 0 into buffer 0
  async_load_b128(&sAh[0][so], Ah + gA);
  async_load_b128(&sAl[0][so], Al + gA);
  async_load_b128(&sBh[0][so], Bh + gB);
  async_load_b128(&sBl[0][so], Bl + gB);
  wait_async0();
  __syncthreads();

  int buf = 0;
  for (int k0 = 0; k0 < K; k0 += 32) {
    if (k0 + 32 < K) {  // stage next chunk into the other buffer (overlaps compute)
      const int nb = buf ^ 1;
      async_load_b128(&sAh[nb][so], Ah + gA + k0 + 32);
      async_load_b128(&sAl[nb][so], Al + gA + k0 + 32);
      async_load_b128(&sBh[nb][so], Bh + gB + k0 + 32);
      async_load_b128(&sBl[nb][so], Bl + gB + k0 + 32);
    }
    {
      const int o0 = hi * 16, o1 = hi * 16 + 8, ob = hi * 16;
      const int ra0 = (wm * 32 + lm) * LS, ra1 = (wm * 32 + 16 + lm) * LS;
      const int rb  = (wn * 16 + lm) * LS;
      v16h a0h = load_frag_2x8(&sAh[buf][ra0 + o0], &sAh[buf][ra0 + o1]);
      v16h a0l = load_frag_2x8(&sAl[buf][ra0 + o0], &sAl[buf][ra0 + o1]);
      v16h a1h = load_frag_2x8(&sAh[buf][ra1 + o0], &sAh[buf][ra1 + o1]);
      v16h a1l = load_frag_2x8(&sAl[buf][ra1 + o0], &sAl[buf][ra1 + o1]);
      v16h bh  = load_frag_16(&sBh[buf][rb + ob]);
      v16h bl  = load_frag_16(&sBl[buf][rb + ob]);
      acc[0] = wmma16(a0h, bh, acc[0]);
      acc[0] = wmma16(a0h, bl, acc[0]);
      acc[0] = wmma16(a0l, bh, acc[0]);
      acc[1] = wmma16(a1h, bh, acc[1]);
      acc[1] = wmma16(a1h, bl, acc[1]);
      acc[1] = wmma16(a1l, bh, acc[1]);
    }
    wait_async0();     // my async writes for the next buffer have landed
    __syncthreads();   // everyone's have; everyone done reading current buffer
    buf ^= 1;
  }

#pragma unroll
  for (int t = 0; t < 2; ++t) {
#pragma unroll
    for (int j = 0; j < 8; ++j) {
      const int row = bm + wm * 32 + t * 16 + hi * 8 + j;  // C/D: M = 8*hi + j
      const int col = bn + wn * 16 + lm;                   //      N = lane%16
      const float v = acc[t][j];
      if (MODE == 0) {
        Cf[(size_t)row * N + col] = v;
      } else {
        const int b = row >> 11, s = row & 2047;
        const int h = col >> 7,  c = col & 127;
        size_t idx;
        if (MODE == 1) idx = (((size_t)b * HEADS + h) * S_LEN + s) * DH + c;
        else           idx = (((size_t)b * HEADS + h) * DH + c) * S_LEN + s;
        const f16 hh = (f16)v;
        Ch[idx] = hh;
        Cl[idx] = (f16)(v - (float)hh);
      }
    }
  }
}

// ---------------------------------------------------------------------------
// Fused attention: per (b,h) and 16-row query block:
//   scores (WMMA) -> +bias +mask -> softmax (LDS, hi/lo f16) -> score out (NT)
//   ctx = P @ V (WMMA, V pre-transposed) -> split-f16 ctx [b,s,h*128]
// LDS: Ph/Pl 16x2056 halves + Qh/Ql 16x136 halves = 140,288 B (2 WGs / WGP).
// ---------------------------------------------------------------------------
#define HST 2056   // padded score row stride (halves)
#define QST 136    // padded Q row stride (halves)
#define ATTN_LDS_BYTES ((2 * 16 * HST + 2 * 16 * QST) * 2)

__global__ __launch_bounds__(256) void attn_fused(
    const f16* __restrict__ qh, const f16* __restrict__ ql,
    const f16* __restrict__ kh, const f16* __restrict__ kl,
    const f16* __restrict__ vth, const f16* __restrict__ vtl,
    const float* __restrict__ bias, const float* __restrict__ mask,
    float* __restrict__ scoreOut, f16* __restrict__ ctxh, f16* __restrict__ ctxl) {
  extern __shared__ char smem[];
  f16* Ph  = (f16*)smem;          // [16][HST]
  f16* Pl  = Ph + 16 * HST;
  f16* Qsh = Pl + 16 * HST;       // [16][QST]
  f16* Qsl = Qsh + 16 * QST;

  const int bh = blockIdx.y;
  const int b = bh >> 4, h = bh & 15;
  const int row0 = blockIdx.x * 16;
  const int tid = threadIdx.x, wave = tid >> 5, lane = tid & 31;
  const int hi = lane >> 4, lm = lane & 15;

  const f16* Qhg = qh + ((size_t)bh * S_LEN + row0) * DH;
  const f16* Qlg = ql + ((size_t)bh * S_LEN + row0) * DH;
  const f16* Khg = kh + (size_t)bh * S_LEN * DH;
  const f16* Klg = kl + (size_t)bh * S_LEN * DH;

  {  // stage Q tile via async Global->LDS: 16x128 halves x2, one b128/thread/array
    const int r = tid >> 4, c8 = (tid & 15) * 8;
    async_load_b128(&Qsh[r * QST + c8], Qhg + r * DH + c8);
    async_load_b128(&Qsl[r * QST + c8], Qlg + r * DH + c8);
  }
  wait_async0();
  __syncthreads();

  // ---- score phase: wave w owns key tiles ct = w, w+8, ... (16 each) ----
  const float scl = 0.08838834764831845f;  // 1/sqrt(128)
  for (int ct = wave; ct < S_LEN / 16; ct += 8) {
    v8f acc = {};
    const f16* KhRow = Khg + (size_t)(ct * 16 + lm) * DH;
    const f16* KlRow = Klg + (size_t)(ct * 16 + lm) * DH;
#pragma unroll
    for (int kc = 0; kc < DH / 32; ++kc) {
      const int o0 = kc * 32 + hi * 16, o1 = kc * 32 + hi * 16 + 8, ob = kc * 32 + hi * 16;
      v16h ah = load_frag_2x8(&Qsh[lm * QST + o0], &Qsh[lm * QST + o1]);
      v16h al = load_frag_2x8(&Qsl[lm * QST + o0], &Qsl[lm * QST + o1]);
      v16h bh2 = load_frag_16(KhRow + ob);
      v16h bl2 = load_frag_16(KlRow + ob);
      acc = wmma16(ah, bh2, acc);
      acc = wmma16(ah, bl2, acc);
      acc = wmma16(al, bh2, acc);
    }
    const int col = ct * 16 + lm;
#pragma unroll
    for (int j = 0; j < 8; ++j) {
      const int r = hi * 8 + j;
      const int grow = row0 + r;
      const float v = acc[j] * scl
                    + bias[((size_t)h * S_LEN + grow) * S_LEN + col]
                    + mask[(size_t)grow * S_LEN + col];
      const f16 hh = (f16)v;
      Ph[r * HST + col] = hh;
      Pl[r * HST + col] = (f16)(v - (float)hh);
    }
  }
  __syncthreads();

  // ---- softmax: 8 waves x 2 rows, wave32 shuffle reductions ----
  float* sOut = scoreOut + ((size_t)bh * S_LEN + row0) * S_LEN;
  for (int r = wave * 2; r < wave * 2 + 2; ++r) {
    float mx = -3.0e38f;
    for (int c = lane; c < S_LEN; c += 32)
      mx = fmaxf(mx, (float)Ph[r * HST + c] + (float)Pl[r * HST + c]);
#pragma unroll
    for (int off = 16; off; off >>= 1) mx = fmaxf(mx, __shfl_xor(mx, off, 32));
    float sum = 0.f;
    for (int c = lane; c < S_LEN; c += 32) {
      const float e = __expf(((float)Ph[r * HST + c] + (float)Pl[r * HST + c]) - mx);
      const f16 eh = (f16)e;
      Ph[r * HST + c] = eh;
      Pl[r * HST + c] = (f16)(e - (float)eh);
      sum += e;
    }
#pragma unroll
    for (int off = 16; off; off >>= 1) sum += __shfl_xor(sum, off, 32);
    const float inv = 1.0f / sum;
    for (int c = lane; c < S_LEN; c += 32) {
      const float pv = ((float)Ph[r * HST + c] + (float)Pl[r * HST + c]) * inv;
      __builtin_nontemporal_store(pv, sOut + (size_t)r * S_LEN + c);  // stream 1GB output
      const f16 phv = (f16)pv;
      Ph[r * HST + c] = phv;
      Pl[r * HST + c] = (f16)(pv - (float)phv);
    }
  }
  __syncthreads();

  // ---- ctx = P @ V : wave w owns output col tile w (8 tiles cover 128) ----
  {
    v8f acc = {};
    const f16* VhRow = vth + ((size_t)bh * DH + wave * 16 + lm) * S_LEN;
    const f16* VlRow = vtl + ((size_t)bh * DH + wave * 16 + lm) * S_LEN;
#pragma unroll 4
    for (int kc = 0; kc < S_LEN / 32; ++kc) {
      const int o0 = kc * 32 + hi * 16, o1 = kc * 32 + hi * 16 + 8, ob = kc * 32 + hi * 16;
      v16h ah = load_frag_2x8(&Ph[lm * HST + o0], &Ph[lm * HST + o1]);
      v16h al = load_frag_2x8(&Pl[lm * HST + o0], &Pl[lm * HST + o1]);
      v16h bh2 = load_frag_16(VhRow + ob);
      v16h bl2 = load_frag_16(VlRow + ob);
      acc = wmma16(ah, bh2, acc);
      acc = wmma16(ah, bl2, acc);
      acc = wmma16(al, bh2, acc);
    }
#pragma unroll
    for (int j = 0; j < 8; ++j) {
      const int r = hi * 8 + j;
      const size_t idx = ((size_t)b * S_LEN + row0 + r) * (HEADS * DH) + h * DH + wave * 16 + lm;
      const float v = acc[j];
      const f16 hh = (f16)v;
      ctxh[idx] = hh;
      ctxl[idx] = (f16)(v - (float)hh);
    }
  }
}

// ---------------------------------------------------------------------------
extern "C" void kernel_launch(void* const* d_in, const int* in_sizes, int n_in,
                              void* d_out, int out_size, void* d_ws, size_t ws_size,
                              hipStream_t stream) {
  (void)in_sizes; (void)n_in; (void)out_size; (void)ws_size;

  const float* x    = (const float*)d_in[0];
  const float* mask = (const float*)d_in[1];
  const float* bias = (const float*)d_in[2];
  const float* wq   = (const float*)d_in[3];
  const float* wk   = (const float*)d_in[4];
  const float* wv   = (const float*)d_in[5];
  const float* wo   = (const float*)d_in[6];

  float* out   = (float*)d_out;
  float* score = out + (size_t)BATCH * S_LEN * 1024;  // outputs: (out, score) concat

  // workspace carve (~336 MB total)
  char* ws = (char*)d_ws;
  auto carve = [&](size_t bytes) { char* p = ws; ws += (bytes + 255) & ~(size_t)255; return p; };
  const size_t XE = (size_t)BATCH * S_LEN * 1024;              // 8,388,608
  const size_t WE = (size_t)2048 * 1024;                       // 2,097,152 (all 4 weights)
  const size_t QE = (size_t)BATCH * HEADS * S_LEN * DH;        // 16,777,216

  f16* xh  = (f16*)carve(XE * 2);  f16* xl  = (f16*)carve(XE * 2);
  f16* wqh = (f16*)carve(WE * 2);  f16* wql = (f16*)carve(WE * 2);
  f16* wkh = (f16*)carve(WE * 2);  f16* wkl = (f16*)carve(WE * 2);
  f16* wvh = (f16*)carve(WE * 2);  f16* wvl = (f16*)carve(WE * 2);
  f16* woh = (f16*)carve(WE * 2);  f16* wol = (f16*)carve(WE * 2);
  f16* qbh = (f16*)carve(QE * 2);  f16* qbl = (f16*)carve(QE * 2);
  f16* kbh = (f16*)carve(QE * 2);  f16* kbl = (f16*)carve(QE * 2);
  f16* vth = (f16*)carve(QE * 2);  f16* vtl = (f16*)carve(QE * 2);
  f16* cxh = (f16*)carve(QE * 2);  f16* cxl = (f16*)carve(QE * 2);

  // 1) hi/lo splits
  split_f32_f16<<<dim3((unsigned)((XE + 255) / 256)), 256, 0, stream>>>(x,  xh,  xl,  (int)XE);
  split_f32_f16<<<dim3((unsigned)((WE + 255) / 256)), 256, 0, stream>>>(wq, wqh, wql, (int)WE);
  split_f32_f16<<<dim3((unsigned)((WE + 255) / 256)), 256, 0, stream>>>(wk, wkh, wkl, (int)WE);
  split_f32_f16<<<dim3((unsigned)((WE + 255) / 256)), 256, 0, stream>>>(wv, wvh, wvl, (int)WE);
  split_f32_f16<<<dim3((unsigned)((WE + 255) / 256)), 256, 0, stream>>>(wo, woh, wol, (int)WE);

  // 2) projections: M=8192, N=2048, K=1024 -> grid (N/64, M/64)
  gemm_split<1><<<dim3(32, 128), 256, 0, stream>>>(xh, xl, wqh, wql, 1024, 2048, nullptr, qbh, qbl);
  gemm_split<1><<<dim3(32, 128), 256, 0, stream>>>(xh, xl, wkh, wkl, 1024, 2048, nullptr, kbh, kbl);
  gemm_split<2><<<dim3(32, 128), 256, 0, stream>>>(xh, xl, wvh, wvl, 1024, 2048, nullptr, vth, vtl);

  // 3) fused attention: grid = (S/16 row blocks, B*H)
  (void)hipFuncSetAttribute((const void*)attn_fused,
                            hipFuncAttributeMaxDynamicSharedMemorySize, ATTN_LDS_BYTES);
  attn_fused<<<dim3(S_LEN / 16, BATCH * HEADS), 256, ATTN_LDS_BYTES, stream>>>(
      qbh, qbl, kbh, kbl, vth, vtl, bias, mask, score, cxh, cxl);

  // 4) out = ctx @ wo^T : M=8192, N=1024, K=2048
  gemm_split<0><<<dim3(16, 128), 256, 0, stream>>>(cxh, cxl, woh, wol, 2048, 1024, out, nullptr, nullptr);
}
